// CustomModel_41824391528833
// MI455X (gfx1250) — compile-verified, matrix operand order checked by reading, not run on hardware
//
#include <hip/hip_runtime.h>

#define CCH 32
#define D1  1024
#define D2  256
#define HID 64
#define HW_TOT (1024 * 1024)
#define PIX_PER_WAVE 64          // 4 A-tiles of 16 pixels per wave
#define WAVES_PER_BLK 8

typedef __attribute__((ext_vector_type(16))) _Float16 v16h;
typedef __attribute__((ext_vector_type(8)))  _Float16 v8h;
typedef __attribute__((ext_vector_type(8)))  float    v8f;

// ---------------- Prologue 1: transpose tables to channel-last ----------------

__global__ __launch_bounds__(256) void tq_transpose_lines(
    const float* __restrict__ lx, const float* __restrict__ ly,
    float* __restrict__ lxT, float* __restrict__ lyT) {
  int idx = blockIdx.x * 256 + threadIdx.x;
  if (idx >= D1 * CCH) return;
  int pos = idx >> 5;
  int c   = idx & 31;
  lxT[idx] = lx[c * D1 + pos];
  lyT[idx] = ly[c * D1 + pos];
}

__global__ __launch_bounds__(256) void tq_transpose_plane(
    const float* __restrict__ pl, float* __restrict__ pT) {
  int idx = blockIdx.x * 256 + threadIdx.x;
  if (idx >= D2 * D2 * CCH) return;
  int rest = idx >> 5;
  int c    = idx & 31;
  pT[idx] = pl[c * (D2 * D2) + rest];
}

// ---------------- Prologue 2: pack weights into WMMA lane layout -------------
// B1 tiles (layer 1, W1^T): Bpk[t][lane][e], t=0..3.
//   lane l: N = t*16 + (l&15), kbase = (l<16)?0:8,
//   e 0..7 -> K = kbase+e ; e 8..15 -> K = kbase+16+(e-8).
// W2 operand (layer 2, used as the A matrix, rows all equal): W2pk[s][lane][e].
//   K-dim uses the PERMUTED hidden order j = 4*n' + t  (orig neuron = (j&3)*16 + (j>>2)),
//   matching how h is laid out in LDS by the fused kernel.

__global__ __launch_bounds__(128) void tq_pack_weights(
    const float* __restrict__ W1, const float* __restrict__ W2,
    _Float16* __restrict__ Bpk, _Float16* __restrict__ W2pk) {
  const int tid   = threadIdx.x;          // 128 threads: 4 tiles x 32 lanes
  const int t     = tid >> 5;
  const int lane  = tid & 31;
  const int nloc  = lane & 15;
  const int kbase = (lane < 16) ? 0 : 8;

  const float* w1r = W1 + (t * 16 + nloc) * CCH;
  _Float16* dst = Bpk + ((t * 32) + lane) * 16;
#pragma unroll
  for (int e = 0; e < 8; ++e) {
    dst[e]     = (_Float16)w1r[kbase + e];
    dst[e + 8] = (_Float16)w1r[kbase + 16 + e];
  }

  if (t < 2) {                            // 2 K-slices of the layer-2 vector
    _Float16* dst2 = W2pk + ((t * 32) + lane) * 16;
#pragma unroll
    for (int e = 0; e < 8; ++e) {
      const int j0 = t * 32 + kbase + e;
      const int j1 = t * 32 + kbase + 16 + e;
      dst2[e]     = (_Float16)W2[(j0 & 3) * 16 + (j0 >> 2)];
      dst2[e + 8] = (_Float16)W2[(j1 & 3) * 16 + (j1 >> 2)];
    }
  }
}

// ---------------- Fused interpolation + 2-layer WMMA MLP ----------------
// 256 threads = 8 wave32; each wave: 64 pixels, 24 WMMAs (16 L1 + 8 L2).

__global__ __launch_bounds__(256) void tq_fused(
    const float* __restrict__ coords,      // [HW][2]
    const float* __restrict__ lxT,         // [D1][C]
    const float* __restrict__ lyT,         // [D1][C]
    const float* __restrict__ pT,          // [D2][D2][C]
    const _Float16* __restrict__ Bpk,      // [4][32][16] packed W1^T tiles
    const _Float16* __restrict__ W2pk,     // [2][32][16] packed W2 (A operand)
    const float* __restrict__ b1,          // [64]
    const float* __restrict__ b2,          // [1]
    float* __restrict__ out)               // [HW]
{
  __shared__ _Float16 feat [WAVES_PER_BLK][PIX_PER_WAVE][CCH];  // 32 KB
  __shared__ _Float16 feat2[WAVES_PER_BLK][16][HID];            // 16 KB, permuted j=4n'+t

  const int lane = threadIdx.x & 31;
  const int wave = threadIdx.x >> 5;
  const int p0   = (blockIdx.x * WAVES_PER_BLK + wave) * PIX_PER_WAVE;

  const int nloc  = lane & 15;
  const int kbase = (lane < 16) ? 0 : 8;
  const int hi8   = (lane >> 4) << 3;     // 0 for lanes 0-15, 8 for 16-31

  // ---- Load pre-packed WMMA operands (12x b128 loads) ----
  v16h Btile[4];
#pragma unroll
  for (int t = 0; t < 4; ++t) {
    const _Float16* src = Bpk + ((t * 32) + lane) * 16;
    const v8h blo = *(const v8h*)(src);
    const v8h bhi = *(const v8h*)(src + 8);
    Btile[t] = __builtin_shufflevector(blo, bhi,
        0, 1, 2, 3, 4, 5, 6, 7, 8, 9, 10, 11, 12, 13, 14, 15);
  }
  v16h A2tile[2];
#pragma unroll
  for (int s = 0; s < 2; ++s) {
    const _Float16* src = W2pk + ((s * 32) + lane) * 16;
    const v8h blo = *(const v8h*)(src);
    const v8h bhi = *(const v8h*)(src + 8);
    A2tile[s] = __builtin_shufflevector(blo, bhi,
        0, 1, 2, 3, 4, 5, 6, 7, 8, 9, 10, 11, 12, 13, 14, 15);
  }
  // Packed f16 biases for (t0,t1) and (t2,t3) at this lane's column.
  unsigned biasLo, biasHi;
  {
    const float bb0 = b1[0 * 16 + nloc], bb1 = b1[1 * 16 + nloc];
    const float bb2 = b1[2 * 16 + nloc], bb3 = b1[3 * 16 + nloc];
    asm("v_cvt_pk_f16_f32 %0, %1, %2" : "=v"(biasLo) : "v"(bb0), "v"(bb1));
    asm("v_cvt_pk_f16_f32 %0, %1, %2" : "=v"(biasHi) : "v"(bb2), "v"(bb3));
  }
  const float b2s = b2[0];
  unsigned zreg;
  asm("v_mov_b32 %0, 0" : "=v"(zreg));    // packed-f16 zero for ReLU
  const v8f Czero = {};                   // literal-0 C operand for WMMA

  // ---- Feature phase: 4 sub-tiles; lane -> (pixel lane/2, channels (lane&1)*16) ----
  const int plx = lane >> 1;
  const int ch0 = (lane & 1) * 16;
  for (int it = 0; it < 4; ++it) {
    const int   p  = p0 + it * 16 + plx;
    const float cx = coords[2 * p + 0];
    const float cy = coords[2 * p + 1];
    const float xn = cx * (2.0f / 1024.0f) - 1.0f;
    const float yn = cy * (2.0f / 1024.0f) - 1.0f;

    // line interp (border clamp, align_corners)
    float posx = fminf(fmaxf((xn + 1.0f) * 0.5f * (float)(D1 - 1), 0.0f), (float)(D1 - 1));
    float posy = fminf(fmaxf((yn + 1.0f) * 0.5f * (float)(D1 - 1), 0.0f), (float)(D1 - 1));
    const int   i0x = min((int)posx, D1 - 2);
    const int   i0y = min((int)posy, D1 - 2);
    const float wlx = posx - (float)i0x;
    const float wly = posy - (float)i0y;

    // plane interp (zeros padding, align_corners)
    const float pxf = (xn + 1.0f) * 0.5f * (float)(D2 - 1);
    const float pyf = (yn + 1.0f) * 0.5f * (float)(D2 - 1);
    const int   x0  = (int)floorf(pxf);
    const int   y0  = (int)floorf(pyf);
    const float wx  = pxf - (float)x0;
    const float wy  = pyf - (float)y0;
    const int x1 = x0 + 1, y1 = y0 + 1;
    const bool vx0 = (x0 >= 0) && (x0 < D2);
    const bool vx1 = (x1 >= 0) && (x1 < D2);
    const bool vy0 = (y0 >= 0) && (y0 < D2);
    const bool vy1 = (y1 >= 0) && (y1 < D2);
    const float w00 = (vy0 && vx0) ? (1.0f - wy) * (1.0f - wx) : 0.0f;
    const float w01 = (vy0 && vx1) ? (1.0f - wy) * wx          : 0.0f;
    const float w10 = (vy1 && vx0) ? wy * (1.0f - wx)          : 0.0f;
    const float w11 = (vy1 && vx1) ? wy * wx                   : 0.0f;
    const int x0c = min(max(x0, 0), D2 - 1), x1c = min(max(x1, 0), D2 - 1);
    const int y0c = min(max(y0, 0), D2 - 1), y1c = min(max(y1, 0), D2 - 1);

    const float4* a0  = (const float4*)(lxT + i0x * CCH + ch0);
    const float4* a1  = (const float4*)(lxT + (i0x + 1) * CCH + ch0);
    const float4* c0  = (const float4*)(lyT + i0y * CCH + ch0);
    const float4* c1  = (const float4*)(lyT + (i0y + 1) * CCH + ch0);
    const float4* q00 = (const float4*)(pT + (y0c * D2 + x0c) * CCH + ch0);
    const float4* q01 = (const float4*)(pT + (y0c * D2 + x1c) * CCH + ch0);
    const float4* q10 = (const float4*)(pT + (y1c * D2 + x0c) * CCH + ch0);
    const float4* q11 = (const float4*)(pT + (y1c * D2 + x1c) * CCH + ch0);

    _Float16* dst = &feat[wave][it * 16 + plx][ch0];
#pragma unroll
    for (int g = 0; g < 4; ++g) {
      const float4 va0 = a0[g],  va1 = a1[g];
      const float4 vc0 = c0[g],  vc1 = c1[g];
      const float4 v00 = q00[g], v01 = q01[g], v10 = q10[g], v11 = q11[g];
#define TQ_COMP(j, fld)                                                        \
      {                                                                        \
        float fxv = va0.fld + wlx * (va1.fld - va0.fld);                       \
        float fyv = vc0.fld + wly * (vc1.fld - vc0.fld);                       \
        float fpv = w00 * v00.fld + w01 * v01.fld                              \
                  + w10 * v10.fld + w11 * v11.fld;                             \
        dst[g * 4 + j] = (_Float16)(fxv * fyv + fpv);                          \
      }
      TQ_COMP(0, x) TQ_COMP(1, y) TQ_COMP(2, z) TQ_COMP(3, w)
#undef TQ_COMP
    }
  }

  __syncthreads();

  // ---- Compute phase: per A-tile: 4 L1 WMMAs + packed epilogue + 2 L2 WMMAs ----
  const int m = lane & 15;
#pragma unroll
  for (int it = 0; it < 4; ++it) {
    // A (16x32 f16): lanes 0-15 hold K 0..7/16..23 of pixel M=lane;
    // lanes 16-31 hold K 8..15/24..31 of pixel M=lane-16.
    const v8h alo = *(const v8h*)&feat[wave][it * 16 + m][kbase];
    const v8h ahi = *(const v8h*)&feat[wave][it * 16 + m][kbase + 16];
    const v16h A  = __builtin_shufflevector(alo, ahi,
        0, 1, 2, 3, 4, 5, 6, 7, 8, 9, 10, 11, 12, 13, 14, 15);

    // Layer 1: W1*feat with literal-0 accumulator (bias added in f16 below).
    v8f acc[4];
#pragma unroll
    for (int t = 0; t < 4; ++t) {
      acc[t] = __builtin_amdgcn_wmma_f32_16x16x32_f16(
          false, A, false, Btile[t], (short)0, Czero, false, false);
    }

    // Epilogue: pack (t0..t3) of each pixel row, +bias, ReLU, one b64 store.
    // Permuted LDS layout: h[pixel][j], j = 4*nloc + t  -> lane's 4 values contiguous.
#pragma unroll
    for (int r = 0; r < 8; ++r) {
      unsigned lo, hi;
      asm("v_cvt_pk_f16_f32 %0, %1, %2" : "=v"(lo) : "v"(acc[0][r]), "v"(acc[1][r]));
      asm("v_cvt_pk_f16_f32 %0, %1, %2" : "=v"(hi) : "v"(acc[2][r]), "v"(acc[3][r]));
      asm("v_pk_add_f16 %0, %1, %2"     : "=v"(lo) : "v"(lo), "v"(biasLo));
      asm("v_pk_add_f16 %0, %1, %2"     : "=v"(hi) : "v"(hi), "v"(biasHi));
      asm("v_pk_max_num_f16 %0, %1, %2" : "=v"(lo) : "v"(lo), "v"(zreg));
      asm("v_pk_max_num_f16 %0, %1, %2" : "=v"(hi) : "v"(hi), "v"(zreg));
      *(uint2*)&feat2[wave][r + hi8][nloc * 4] = make_uint2(lo, hi);
    }

    // Layer 2: A = W2 broadcast (rows equal), B = h (columns = pixels).
    // D[m][n] = out[pixel n] for every row m.
    v8f acc2;
#pragma unroll
    for (int s = 0; s < 2; ++s) {
      const v8h blo2 = *(const v8h*)&feat2[wave][nloc][s * 32 + kbase];
      const v8h bhi2 = *(const v8h*)&feat2[wave][nloc][s * 32 + kbase + 16];
      const v16h Bh  = __builtin_shufflevector(blo2, bhi2,
          0, 1, 2, 3, 4, 5, 6, 7, 8, 9, 10, 11, 12, 13, 14, 15);
      acc2 = __builtin_amdgcn_wmma_f32_16x16x32_f16(
          false, A2tile[s], false, Bh, (short)0, (s == 0) ? Czero : acc2,
          false, false);
    }

    // Single store: lanes 0-15 each own one pixel (column), any row works.
    if (lane < 16) {
      out[p0 + it * 16 + nloc] = acc2[0] + b2s;
    }
  }
}

// ---------------- Host-side launch ----------------

extern "C" void kernel_launch(void* const* d_in, const int* in_sizes, int n_in,
                              void* d_out, int out_size, void* d_ws, size_t ws_size,
                              hipStream_t stream) {
  (void)in_sizes; (void)n_in; (void)out_size; (void)ws_size;

  const float* coords = (const float*)d_in[0];
  const float* lx     = (const float*)d_in[1];
  const float* ly     = (const float*)d_in[2];
  const float* plane  = (const float*)d_in[3];
  const float* W1     = (const float*)d_in[4];
  const float* b1     = (const float*)d_in[5];
  const float* W2     = (const float*)d_in[6];
  const float* b2     = (const float*)d_in[7];
  float* out = (float*)d_out;

  // Workspace: channel-last tables (8.25 MB) + packed operand tiles (6 KB)
  float* lxT = (float*)d_ws;
  float* lyT = lxT + D1 * CCH;
  float* pT  = lyT + D1 * CCH;
  _Float16* Bpk  = (_Float16*)(pT + D2 * D2 * CCH);
  _Float16* W2pk = Bpk + 4 * 32 * 16;

  tq_transpose_lines<<<(D1 * CCH + 255) / 256, 256, 0, stream>>>(lx, ly, lxT, lyT);
  tq_transpose_plane<<<(D2 * D2 * CCH + 255) / 256, 256, 0, stream>>>(plane, pT);
  tq_pack_weights<<<1, 128, 0, stream>>>(W1, W2, Bpk, W2pk);

  // 1M pixels, 64 px per wave, 8 waves per block -> 2048 blocks
  tq_fused<<<HW_TOT / (PIX_PER_WAVE * WAVES_PER_BLK), 256, 0, stream>>>(
      coords, lxT, lyT, pT, Bpk, W2pk, b1, b2, out);
}
